// PSFGNNAdvanced_29918742184025
// MI455X (gfx1250) — compile-verified, matrix operand order checked by reading, not run on hardware
//
#include <hip/hip_runtime.h>
#include <math.h>

// ---------------- sizes (match reference) ----------------
#define HEADS 6

// ---------------- bf16 / WMMA types ----------------
typedef __bf16 bf16;
typedef bf16  v16bf __attribute__((ext_vector_type(16)));
typedef bf16  v8bf  __attribute__((ext_vector_type(8)));
typedef float v8f   __attribute__((ext_vector_type(8)));

static __device__ __forceinline__ unsigned short bfbits(float f) {
  union { float f; unsigned u; } c; c.f = f;
  return (unsigned short)((c.u + 0x7FFFu + ((c.u >> 16) & 1u)) >> 16);  // RNE
}
static __device__ __forceinline__ bf16 f2bf(float f) {
  unsigned short s = bfbits(f);
  bf16 b; __builtin_memcpy(&b, &s, 2);
  return b;
}
static __device__ __forceinline__ unsigned pack2(float a, float b) {
  return (unsigned)bfbits(a) | ((unsigned)bfbits(b) << 16);
}

// =====================================================================
// Generic GEMM:  C[M,N] = act(A[M,K] @ W[K,N] + bias)
// bf16 staged via LDS, fp32 WMMA accumulate.
// Block = 256 threads = 8 waves; 64x64 macro-tile; each wave owns two
// 16x16 tiles sharing one A fragment (2 v_wmma per K-step).
// LDS layouts chosen so each fragment = two contiguous 8xbf16 runs
// (ISA 7.12.2): A row-major As[m][k], B transposed Bt[n][k] ->
// two ds_load_b128 per fragment.
// Full-interior tiles stage with float4 (global_load_b128) + packed
// bf16x4 LDS stores; edge tiles fall back to bounds-checked scalars.
// =====================================================================
#define BM 64
#define BN 64
#define BK 32

__global__ __launch_bounds__(256) void gemm_wmma_k(
    const float* __restrict__ A, const float* __restrict__ W,
    const float* __restrict__ bias, float* __restrict__ C,
    int M, int K, int N, int act)
{
  __shared__ bf16 As[BM][BK];   // 4KB, 64B rows (16B-aligned chunks)
  __shared__ bf16 Bt[BN][BK];   // 4KB, B transposed: Bt[n][k]

  const int tid  = threadIdx.x;
  const int lane = tid & 31;
  const int wv   = tid >> 5;          // 0..7
  const int wm   = wv >> 1;           // 0..3 row tile
  const int wcol = (wv & 1) * 32;     // 0 or 32: column-pair base
  const int half = lane >> 4;         // 0/1 -> k-base 0/8
  const int r    = lane & 15;
  const int row0 = blockIdx.y * BM;
  const int col0 = blockIdx.x * BN;

  const bool vecOK_A = ((K & 3) == 0) && (row0 + BM <= M);
  const bool vecOK_B = ((N & 3) == 0) && (col0 + BN <= N);

  v8f acc0 = {0.f,0.f,0.f,0.f,0.f,0.f,0.f,0.f};
  v8f acc1 = {0.f,0.f,0.f,0.f,0.f,0.f,0.f,0.f};

  for (int k0 = 0; k0 < K; k0 += BK) {
    const bool fullK = (k0 + BK) <= K;

    // ---- stage A (64x32) ----
    if (vecOK_A && fullK) {
      // 512 float4 loads (global_load_b128), packed bf16x4 LDS stores
      for (int i = tid; i < (BM * BK) / 4; i += 256) {
        int m = i >> 3, kv = i & 7;                 // 8 float4 per row
        const float4* Ar = (const float4*)(A + (size_t)(row0 + m) * K + k0);
        float4 f = Ar[kv];
        uint2 p; p.x = pack2(f.x, f.y); p.y = pack2(f.z, f.w);
        *(uint2*)&As[m][kv * 4] = p;
        if (kv == 0 && (k0 + BK) < K)
          __builtin_prefetch(&A[(size_t)(row0 + m) * K + k0 + BK], 0, 3);
      }
    } else {
      for (int i = tid; i < BM * BK; i += 256) {
        int m = i >> 5, k = i & 31;
        int gr = row0 + m, gc = k0 + k;
        float v = (gr < M && gc < K) ? A[(size_t)gr * K + gc] : 0.f;
        As[m][k] = f2bf(v);
      }
    }
    // ---- stage B transposed (read coalesced along n, scatter to Bt[n][k]) ----
    if (vecOK_B && fullK) {
      for (int i = tid; i < (BK * BN) / 4; i += 256) {
        int k = i >> 4, nv = i & 15;                // 16 float4 per row
        const float4* Wr = (const float4*)(W + (size_t)(k0 + k) * N + col0);
        float4 f = Wr[nv];
        int n = nv * 4;
        Bt[n + 0][k] = f2bf(f.x);
        Bt[n + 1][k] = f2bf(f.y);
        Bt[n + 2][k] = f2bf(f.z);
        Bt[n + 3][k] = f2bf(f.w);
      }
    } else {
      for (int i = tid; i < BK * BN; i += 256) {
        int k = i >> 6, n = i & 63;
        int gr = k0 + k, gc = col0 + n;
        float v = (gr < K && gc < N) ? W[(size_t)gr * N + gc] : 0.f;
        Bt[n][k] = f2bf(v);
      }
    }
    __syncthreads();

    // ---- vector fragment loads: 2x ds_load_b128 each ----
    const v8bf* ap = (const v8bf*)As[wm * 16 + r];
    v8bf alo = ap[half];          // k = half*8 .. +7
    v8bf ahi = ap[half + 2];      // k = 16 + half*8 .. +7
    v16bf a = __builtin_shufflevector(alo, ahi,
        0,1,2,3,4,5,6,7,8,9,10,11,12,13,14,15);

    const v8bf* bp0 = (const v8bf*)Bt[wcol + r];
    v8bf b0lo = bp0[half], b0hi = bp0[half + 2];
    v16bf b0 = __builtin_shufflevector(b0lo, b0hi,
        0,1,2,3,4,5,6,7,8,9,10,11,12,13,14,15);

    const v8bf* bp1 = (const v8bf*)Bt[wcol + 16 + r];
    v8bf b1lo = bp1[half], b1hi = bp1[half + 2];
    v16bf b1 = __builtin_shufflevector(b1lo, b1hi,
        0,1,2,3,4,5,6,7,8,9,10,11,12,13,14,15);

    acc0 = __builtin_amdgcn_wmma_f32_16x16x32_bf16(false, a, false, b0,
                                                   (short)0, acc0, false, false);
    acc1 = __builtin_amdgcn_wmma_f32_16x16x32_bf16(false, a, false, b1,
                                                   (short)0, acc1, false, false);
    __syncthreads();
  }

  // ---- store (C/D layout: lane halves hold M=r and M=r+8) ----
  int n0 = col0 + wcol + r;
  int n1 = n0 + 16;
#pragma unroll
  for (int t = 0; t < 2; ++t) {
    int n = t ? n1 : n0;
    if (n >= N) continue;
    const v8f& acc = t ? acc1 : acc0;
    float bv = bias ? bias[n] : 0.f;
#pragma unroll
    for (int rr = 0; rr < 8; ++rr) {
      int m = row0 + wm * 16 + rr + half * 8;
      if (m < M) {
        float v = acc[rr] + bv;
        if (act == 1)      v = fmaxf(v, 0.f);
        else if (act == 2) v = 1.f / (1.f + expf(-v));
        C[(size_t)m * N + n] = v;
      }
    }
  }
}

// =====================================================================
// Async-memory probe: GLOBAL_LOAD_ASYNC_TO_LDS_B32 + s_wait_asynccnt
// (ISA 15.18.3). Copies 64 floats global->LDS->global scratch.
// =====================================================================
__global__ __launch_bounds__(64) void async_probe_k(const float* __restrict__ g,
                                                    float* __restrict__ out)
{
  __shared__ float buf[64];
  unsigned lds = (unsigned)(size_t)&buf[threadIdx.x];
  unsigned long long ga = (unsigned long long)(size_t)(g + threadIdx.x);
  asm volatile("global_load_async_to_lds_b32 %0, %1, off"
               :: "v"(lds), "v"(ga) : "memory");
  asm volatile("s_wait_asynccnt 0x0" ::: "memory");
  __syncthreads();
  out[threadIdx.x] = buf[threadIdx.x];
}

// =====================================================================
// LayerNorm: one wave32 per row, shfl_xor reductions. Optional ReLU.
// =====================================================================
__global__ __launch_bounds__(256) void ln_k(
    const float* __restrict__ X, const float* __restrict__ g,
    const float* __restrict__ b, float* __restrict__ Y,
    int M, int D, int relu_after)
{
  int row  = blockIdx.x * 8 + (threadIdx.x >> 5);
  if (row >= M) return;
  int lane = threadIdx.x & 31;
  const float* x = X + (size_t)row * D;
  float s = 0.f;
  for (int i = lane; i < D; i += 32) s += x[i];
  for (int o = 16; o; o >>= 1) s += __shfl_xor(s, o, 32);
  float mean = s / (float)D;
  float v = 0.f;
  for (int i = lane; i < D; i += 32) { float d = x[i] - mean; v += d * d; }
  for (int o = 16; o; o >>= 1) v += __shfl_xor(v, o, 32);
  float rstd = rsqrtf(v / (float)D + 1e-5f);
  float* y = Y + (size_t)row * D;
  for (int i = lane; i < D; i += 32) {
    float t = (x[i] - mean) * rstd * g[i] + b[i];
    if (relu_after) t = fmaxf(t, 0.f);
    y[i] = t;
  }
}

// =====================================================================
// Elementwise / scatter helpers
// =====================================================================
__global__ void fill_k(float* p, long long n, float v) {
  long long i = (long long)blockIdx.x * 256 + threadIdx.x;
  if (i < n) p[i] = v;
}

__global__ void deg_k(const int* __restrict__ idx, float* deg, int E) {
  int i = blockIdx.x * 256 + threadIdx.x;
  if (i < E) atomicAdd(&deg[idx[i]], 1.f);
}

__global__ void node_enc_k(const float* __restrict__ mass, const float* __restrict__ deg,
                           const float* __restrict__ pos,
                           const float* __restrict__ Wm, const float* __restrict__ bm,
                           const float* __restrict__ Wd, const float* __restrict__ bd,
                           float* __restrict__ h0, int N)
{
  long long i = (long long)blockIdx.x * 256 + threadIdx.x;
  if (i >= (long long)N * 64) return;
  int c = (int)(i & 63); int n = (int)(i >> 6);
  float v;
  if (c < 16)       v = mass[n] * Wm[c] + bm[c];
  else if (c < 32)  v = deg[n] * Wd[c - 16] + bd[c - 16];
  else              v = pos[(size_t)n * 32 + (c - 32)];
  h0[i] = v;
}

__global__ void ef_k(float* __restrict__ ef, const float* __restrict__ ea,
                     const float* __restrict__ demb, long long n) {
  long long i = (long long)blockIdx.x * 256 + threadIdx.x;
  if (i >= n) return;
  int c = (int)(i % 32); long long e = i / 32;
  ef[i] = ef[i] * expf(-ea[e] * 0.1f) + demb[c];
}

__global__ void append_col_k(const float* __restrict__ X, const float* __restrict__ colv,
                             float* __restrict__ Y, int M, int D) {
  long long i = (long long)blockIdx.x * 256 + threadIdx.x;
  long long tot = (long long)M * (D + 1);
  if (i >= tot) return;
  int c = (int)(i % (D + 1)); long long m = i / (D + 1);
  Y[i] = (c < D) ? X[m * D + c] : colv[m];
}

__global__ void dot_head_k(const float* __restrict__ X, const float* __restrict__ att,
                           int astride, int aoff, float* __restrict__ out,
                           long long M, int C) {
  long long i = (long long)blockIdx.x * 256 + threadIdx.x;
  if (i >= M * HEADS) return;
  int h = (int)(i % HEADS); long long m = i / HEADS;
  const float* xp = X + m * (long long)HEADS * C + (long long)h * C;
  const float* ap = att + (long long)h * astride + aoff;
  float s = 0.f;
  for (int c = 0; c < C; ++c) s += xp[c] * ap[c];
  out[i] = s;
}

__global__ void alpha_mass_k(const float* __restrict__ ai, const float* __restrict__ aj,
                             const float* __restrict__ ae, const float* __restrict__ mass,
                             const float* __restrict__ Wm, const float* __restrict__ bm,
                             const int* __restrict__ src, const int* __restrict__ dst,
                             float* __restrict__ alpha, long long E) {
  long long i = (long long)blockIdx.x * 256 + threadIdx.x;
  if (i >= E * HEADS) return;
  int h = (int)(i % HEADS); long long e = i / HEADS;
  int s = src[e], d = dst[e];
  float v = ai[(long long)d * HEADS + h] + aj[(long long)s * HEADS + h] + ae[i]
          + mass[d] * Wm[h] + mass[s] * Wm[HEADS + h] + bm[h];
  alpha[i] = (v > 0.f) ? v : 0.2f * v;
}

static __device__ __forceinline__ void atomicMaxF(float* addr, float v) {
  if (v >= 0.f) atomicMax((int*)addr, __float_as_int(v));
  else          atomicMin((unsigned int*)addr, __float_as_uint(v));
}

__global__ void segmax_k(const float* __restrict__ alpha, const int* __restrict__ idx,
                         float* __restrict__ m, long long E) {
  long long i = (long long)blockIdx.x * 256 + threadIdx.x;
  if (i >= E * HEADS) return;
  int h = (int)(i % HEADS); long long e = i / HEADS;
  atomicMaxF(&m[(long long)idx[e] * HEADS + h], alpha[i]);
}

__global__ void expsum_k(float* __restrict__ alpha, const int* __restrict__ idx,
                         const float* __restrict__ m, float* __restrict__ s, long long E) {
  long long i = (long long)blockIdx.x * 256 + threadIdx.x;
  if (i >= E * HEADS) return;
  int h = (int)(i % HEADS); long long e = i / HEADS;
  float mm = m[(long long)idx[e] * HEADS + h];
  if (mm < -3.0e38f) mm = 0.f;                     // where(isfinite(m), m, 0)
  float a = expf(alpha[i] - mm);
  alpha[i] = a;
  atomicAdd(&s[(long long)idx[e] * HEADS + h], a);
}

__global__ void segdiv_k(float* __restrict__ alpha, const int* __restrict__ idx,
                         const float* __restrict__ s, long long E) {
  long long i = (long long)blockIdx.x * 256 + threadIdx.x;
  if (i >= E * HEADS) return;
  int h = (int)(i % HEADS); long long e = i / HEADS;
  alpha[i] = alpha[i] / (s[(long long)idx[e] * HEADS + h] + 1e-16f);
}

__global__ void msg_scatter_k(const float* __restrict__ xs, const float* __restrict__ efp,
                              const float* __restrict__ alpha,
                              const int* __restrict__ src, const int* __restrict__ dst,
                              float* __restrict__ out, long long E, int C) {
  long long i = (long long)blockIdx.x * 256 + threadIdx.x;
  long long tot = E * HEADS * C;
  if (i >= tot) return;
  int c = (int)(i % C); int h = (int)((i / C) % HEADS); long long e = i / ((long long)C * HEADS);
  int HC = HEADS * C;
  float a = alpha[e * HEADS + h];
  float v = (xs[(long long)src[e] * HC + h * C + c] + efp[e * (long long)HC + h * C + c]) * a;
  atomicAdd(&out[(long long)dst[e] * HC + h * C + c], v);
}

__global__ void phys_k(const float* __restrict__ g, const float* __restrict__ t,
                       const float* __restrict__ h, float* __restrict__ o, long long n) {
  long long i = (long long)blockIdx.x * 256 + threadIdx.x;
  if (i >= n) return;
  float gg = g[i];
  o[i] = gg * t[i] + (1.f - gg) * h[i];
}

__global__ void eef_k(const float* __restrict__ ef, const float* __restrict__ sp,
                      const float* __restrict__ mg, const int* __restrict__ src,
                      const int* __restrict__ dst, float* __restrict__ ea2, long long E) {
  long long i = (long long)blockIdx.x * 256 + threadIdx.x;
  if (i >= E * 44) return;
  int c = (int)(i % 44); long long e = i / 44;
  float v;
  if (c < 32)      v = ef[e * 32 + c];
  else if (c < 38) v = sp[(long long)src[e] * HEADS + (c - 32)];
  else             v = mg[(long long)dst[e] * HEADS + (c - 38)];
  ea2[i] = v;
}

__global__ void loopsum_k(const float* __restrict__ ea2, const int* __restrict__ dst,
                          float* __restrict__ loop, long long E) {
  long long i = (long long)blockIdx.x * 256 + threadIdx.x;
  if (i >= E * 44) return;
  int c = (int)(i % 44); long long e = i / 44;
  atomicAdd(&loop[(long long)dst[e] * 44 + c], ea2[e * 44 + c]);
}

__global__ void loopdiv_k(float* __restrict__ loop, const float* __restrict__ degin, int N) {
  long long i = (long long)blockIdx.x * 256 + threadIdx.x;
  if (i >= (long long)N * 44) return;
  long long n = i / 44;
  loop[i] = loop[i] / fmaxf(degin[n], 1.f);
}

__global__ void mkidx_k(const int* __restrict__ src, const int* __restrict__ dst,
                        int* __restrict__ s2, int* __restrict__ d2, int E, int N) {
  int i = blockIdx.x * 256 + threadIdx.x;
  if (i >= E + N) return;
  s2[i] = (i < E) ? src[i] : (i - E);
  d2[i] = (i < E) ? dst[i] : (i - E);
}

__global__ void alpha3_k(const float* __restrict__ as_, const float* __restrict__ ad_,
                         const float* __restrict__ ae_, const int* __restrict__ s2,
                         const int* __restrict__ d2, float* __restrict__ alpha, long long E2) {
  long long i = (long long)blockIdx.x * 256 + threadIdx.x;
  if (i >= E2 * HEADS) return;
  int h = (int)(i % HEADS); long long e = i / HEADS;
  float v = as_[(long long)s2[e] * HEADS + h] + ad_[(long long)d2[e] * HEADS + h] + ae_[i];
  alpha[i] = (v > 0.f) ? v : 0.2f * v;
}

__global__ void scatter3_k(const float* __restrict__ xg, const float* __restrict__ alpha,
                           const int* __restrict__ s2, const int* __restrict__ d2,
                           float* __restrict__ out, long long E2, int C) {
  long long i = (long long)blockIdx.x * 256 + threadIdx.x;
  long long tot = E2 * HEADS * C;
  if (i >= tot) return;
  int c = (int)(i % C); int h = (int)((i / C) % HEADS); long long e = i / ((long long)C * HEADS);
  int HC = HEADS * C;
  float v = xg[(long long)s2[e] * HC + h * C + c] * alpha[e * HEADS + h];
  atomicAdd(&out[(long long)d2[e] * HC + h * C + c], v);
}

__global__ void addrow_k(float* __restrict__ X, const float* __restrict__ b,
                         long long M, int D) {
  long long i = (long long)blockIdx.x * 256 + threadIdx.x;
  if (i >= M * D) return;
  X[i] += b[i % D];
}

__global__ void pool_k(const float* __restrict__ X, int M, int D, float* __restrict__ hg) {
  int w = threadIdx.x >> 5, lane = threadIdx.x & 31;
  for (int c = blockIdx.x * 8 + w; c < D; c += gridDim.x * 8) {
    float s = 0.f, mx = -3.402823466e38f;
    for (int r = lane; r < M; r += 32) {
      float v = X[(size_t)r * D + c];
      s += v; mx = fmaxf(mx, v);
    }
    for (int o = 16; o; o >>= 1) {
      s  += __shfl_xor(s, o, 32);
      mx  = fmaxf(mx, __shfl_xor(mx, o, 32));
    }
    if (lane == 0) { hg[c] = s / (float)M; hg[D + c] = mx; hg[2 * D + c] = s; }
  }
}

__global__ void hf_k(const float* __restrict__ h3, const float* __restrict__ gate,
                     const float* __restrict__ mass, float* __restrict__ hf, int N) {
  long long i = (long long)blockIdx.x * 256 + threadIdx.x;
  if (i >= (long long)N * 65) return;
  int c = (int)(i % 65); long long n = i / 65;
  hf[i] = (c < 64) ? h3[n * 64 + c] * gate[c] : mass[n];
}

__global__ void logsoftmax3_k(float* __restrict__ o, int N) {
  int r = blockIdx.x * 256 + threadIdx.x;
  if (r >= N) return;
  float a = o[r * 3], b = o[r * 3 + 1], c = o[r * 3 + 2];
  float m = fmaxf(a, fmaxf(b, c));
  float l = logf(expf(a - m) + expf(b - m) + expf(c - m)) + m;
  o[r * 3] = a - l; o[r * 3 + 1] = b - l; o[r * 3 + 2] = c - l;
}

// =====================================================================
// Host orchestration
// =====================================================================
static inline int cdiv(long long a, long long b) { return (int)((a + b - 1) / b); }

extern "C" void kernel_launch(void* const* d_in, const int* in_sizes, int n_in,
                              void* d_out, int out_size, void* d_ws, size_t ws_size,
                              hipStream_t stream)
{
  const int N = in_sizes[0];          // x is (N,1)
  const int E = in_sizes[2];          // edge_attr is (E,1)

  const float* x   = (const float*)d_in[0];   // mass column
  const int*   ei  = (const int*)d_in[1];
  const float* ea  = (const float*)d_in[2];
  const int* src = ei;
  const int* dst = ei + E;
#define PF(i) ((const float*)d_in[(i)])

  // ---- workspace bump allocator ----
  char* wsb = (char*)d_ws;
  size_t off = 0;
  auto allocf = [&](long long nf) -> float* {
    float* p = (float*)(wsb + off);
    off += (size_t)(((nf * 4 + 255) / 256) * 256);
    return p;
  };
  auto alloci = [&](long long ni) -> int* {
    int* p = (int*)(wsb + off);
    off += (size_t)(((ni * 4 + 255) / 256) * 256);
    return p;
  };

  float* deg   = allocf(N);
  float* degin = allocf(N);
  float* h0    = allocf((long long)N * 64);
  float* h     = allocf((long long)N * 64);
  float* h1p   = allocf((long long)N * 64);
  float* h2p   = allocf((long long)N * 128);
  float* h3f   = allocf((long long)N * 64);
  float* gateb = allocf((long long)N * 128);
  float* trb   = allocf((long long)N * 128);
  float* xm    = allocf((long long)N * 129);
  float* xs    = allocf((long long)N * 768);
  float* xd    = allocf((long long)N * 768);
  float* hagg  = allocf((long long)N * 768);
  float* hln   = allocf((long long)N * 768);
  float* e1    = allocf((long long)E * 64);
  float* e2    = allocf((long long)E * 128);
  float* ef    = allocf((long long)E * 32);
  float* ebig  = allocf((long long)E * 768);   // also reused as eg ((E+N)*384)
  float* alpha = allocf((long long)(E + N) * HEADS);
  float* aiB   = allocf((long long)N * HEADS);
  float* ajB   = allocf((long long)N * HEADS);
  float* aeB   = allocf((long long)(E + N) * HEADS);
  float* mbuf  = allocf((long long)N * HEADS);
  float* sbuf  = allocf((long long)N * HEADS);
  float* spb   = allocf((long long)N * HEADS);
  float* mgb   = allocf((long long)N * HEADS);
  float* ea2   = allocf((long long)(E + N) * 44);
  float* hg    = allocf(192);
  float* gate64= allocf(64);
  float* aprob = allocf(64);
  int*   s2    = alloci(E + N);
  int*   d2    = alloci(E + N);
  (void)ws_size; (void)n_in;

  auto gemm = [&](const float* A, int K, const float* W, const float* bias,
                  float* C, int M, int Nc, int act) {
    dim3 g(cdiv(Nc, BN), cdiv(M, BM));
    gemm_wmma_k<<<g, 256, 0, stream>>>(A, W, bias, C, M, K, Nc, act);
  };
  auto ln = [&](const float* X, const float* g, const float* b, float* Y,
                int M, int D, int relu) {
    ln_k<<<cdiv(M, 8), 256, 0, stream>>>(X, g, b, Y, M, D, relu);
  };
  auto fill = [&](float* p, long long n, float v) {
    fill_k<<<cdiv(n, 256), 256, 0, stream>>>(p, n, v);
  };
#define EW(kern, total, ...) kern<<<cdiv((long long)(total), 256), 256, 0, stream>>>(__VA_ARGS__)

  const float NEG_INF = -__builtin_huge_valf();

  // async-memory probe (deterministic: copies x[0:64] into scratch)
  async_probe_k<<<1, 64, 0, stream>>>(x, aprob);

  // ---------- node encoder ----------
  fill(deg, N, 0.f);
  EW(deg_k, E, src, deg, E);
  EW(node_enc_k, (long long)N * 64, x, deg, PF(3), PF(4), PF(5), PF(6), PF(7), h0, N);
  gemm(h0, 64, PF(8), PF(9), hagg, N, 64, 0);          // in_proj
  ln(hagg, PF(10), PF(11), h, N, 64, 1);               // in_norm + relu

  // ---------- edge MLP ----------
  gemm(ea, 1, PF(12), PF(13), e1, E, 64, 1);           // ee1 + relu
  ln(e1, PF(14), PF(15), e1, E, 64, 0);                // een1
  gemm(e1, 64, PF(16), PF(17), e2, E, 128, 1);         // ee2 + relu
  ln(e2, PF(18), PF(19), e2, E, 128, 0);               // een2
  gemm(e2, 128, PF(20), PF(21), ef, E, 32, 0);         // ee3
  EW(ef_k, (long long)E * 32, ef, ea, PF(22), (long long)E * 32);

  // ---------- GAT layer 1 (C=64) ----------
  EW(append_col_k, (long long)N * 65, h, x, xm, N, 64);
  gemm(xm, 65, PF(23), PF(24), xs, N, 384, 0);         // c1.src
  gemm(xm, 65, PF(25), PF(26), xd, N, 384, 0);         // c1.dst
  gemm(ef, 32, PF(27), PF(28), ebig, E, 384, 0);       // c1.edge
  EW(dot_head_k, (long long)N * HEADS, xd, PF(29), 192, 0,   aiB, (long long)N, 64);
  EW(dot_head_k, (long long)N * HEADS, xs, PF(29), 192, 64,  ajB, (long long)N, 64);
  EW(dot_head_k, (long long)E * HEADS, ebig, PF(29), 192, 128, aeB, (long long)E, 64);
  EW(alpha_mass_k, (long long)E * HEADS, aiB, ajB, aeB, x, PF(30), PF(31), src, dst, alpha, (long long)E);
  fill(mbuf, (long long)N * HEADS, NEG_INF);
  EW(segmax_k, (long long)E * HEADS, alpha, dst, mbuf, (long long)E);
  fill(sbuf, (long long)N * HEADS, 0.f);
  EW(expsum_k, (long long)E * HEADS, alpha, dst, mbuf, sbuf, (long long)E);
  EW(segdiv_k, (long long)E * HEADS, alpha, dst, sbuf, (long long)E);
  fill(hagg, (long long)N * 384, 0.f);
  EW(msg_scatter_k, (long long)E * HEADS * 64, xs, ebig, alpha, src, dst, hagg, (long long)E, 64);
  ln(hagg, PF(32), PF(33), hln, N, 384, 1);            // n1 + relu
  gemm(hln, 384, PF(34), PF(35), h1p, N, 64, 0);       // p1
  // physics 1
  EW(append_col_k, (long long)N * 65, h1p, x, xm, N, 64);
  gemm(xm, 65, PF(36), PF(37), gateb, N, 64, 2);       // ph1.gate sigmoid
  gemm(h1p, 64, PF(38), PF(39), trb, N, 64, 0);        // ph1.tr
  EW(phys_k, (long long)N * 64, gateb, trb, h1p, h1p, (long long)N * 64);

  // ---------- GAT layer 2 (C=128) ----------
  EW(append_col_k, (long long)N * 65, h1p, x, xm, N, 64);
  gemm(xm, 65, PF(40), PF(41), xs, N, 768, 0);         // c2.src
  gemm(xm, 65, PF(42), PF(43), xd, N, 768, 0);         // c2.dst
  gemm(ef, 32, PF(44), PF(45), ebig, E, 768, 0);       // c2.edge
  EW(dot_head_k, (long long)N * HEADS, xd, PF(46), 384, 0,   aiB, (long long)N, 128);
  EW(dot_head_k, (long long)N * HEADS, xs, PF(46), 384, 128, ajB, (long long)N, 128);
  EW(dot_head_k, (long long)E * HEADS, ebig, PF(46), 384, 256, aeB, (long long)E, 128);
  EW(alpha_mass_k, (long long)E * HEADS, aiB, ajB, aeB, x, PF(47), PF(48), src, dst, alpha, (long long)E);
  fill(mbuf, (long long)N * HEADS, NEG_INF);
  EW(segmax_k, (long long)E * HEADS, alpha, dst, mbuf, (long long)E);
  fill(sbuf, (long long)N * HEADS, 0.f);
  EW(expsum_k, (long long)E * HEADS, alpha, dst, mbuf, sbuf, (long long)E);
  EW(segdiv_k, (long long)E * HEADS, alpha, dst, sbuf, (long long)E);
  fill(hagg, (long long)N * 768, 0.f);
  EW(msg_scatter_k, (long long)E * HEADS * 128, xs, ebig, alpha, src, dst, hagg, (long long)E, 128);
  ln(hagg, PF(49), PF(50), hln, N, 768, 1);            // n2 + relu
  gemm(hln, 768, PF(51), PF(52), h2p, N, 128, 0);      // p2
  // physics 2
  EW(append_col_k, (long long)N * 129, h2p, x, xm, N, 128);
  gemm(xm, 129, PF(53), PF(54), gateb, N, 128, 2);     // ph2.gate sigmoid
  gemm(h2p, 128, PF(55), PF(56), trb, N, 128, 0);      // ph2.tr
  EW(phys_k, (long long)N * 128, gateb, trb, h2p, h2p, (long long)N * 128);

  // ---------- sp / mg heads ----------
  EW(append_col_k, (long long)N * 129, h2p, x, xm, N, 128);   // h2m
  gemm(xm, 129, PF(57), PF(58), hln, N, 64, 1);        // sp1 + relu
  gemm(hln, 64, PF(59), PF(60), spb, N, HEADS, 0);     // sp2
  gemm(xm, 129, PF(61), PF(62), hln, N, 64, 1);        // mg1 + relu
  gemm(hln, 64, PF(63), PF(64), mgb, N, HEADS, 0);     // mg2

  // ---------- eef + loop attr ----------
  EW(eef_k, (long long)E * 44, ef, spb, mgb, src, dst, ea2, (long long)E);
  fill(degin, N, 0.f);
  EW(deg_k, E, dst, degin, E);
  fill(ea2 + (long long)E * 44, (long long)N * 44, 0.f);
  EW(loopsum_k, (long long)E * 44, ea2, dst, ea2 + (long long)E * 44, (long long)E);
  EW(loopdiv_k, (long long)N * 44, ea2 + (long long)E * 44, degin, N);
  EW(mkidx_k, E + N, src, dst, s2, d2, E, N);

  // ---------- GAT layer 3 (self-loops, C=64) ----------
  gemm(h2p, 128, PF(65), (const float*)nullptr, xs, N, 384, 0);          // c3.lin -> xg
  gemm(ea2, 44, PF(66), (const float*)nullptr, ebig, E + N, 384, 0);     // c3.lin_edge -> eg
  EW(dot_head_k, (long long)N * HEADS, xs, PF(67), 64, 0, aiB, (long long)N, 64);       // a_s
  EW(dot_head_k, (long long)N * HEADS, xs, PF(68), 64, 0, ajB, (long long)N, 64);       // a_d
  EW(dot_head_k, (long long)(E + N) * HEADS, ebig, PF(69), 64, 0, aeB, (long long)(E + N), 64);
  EW(alpha3_k, (long long)(E + N) * HEADS, aiB, ajB, aeB, s2, d2, alpha, (long long)(E + N));
  fill(mbuf, (long long)N * HEADS, NEG_INF);
  EW(segmax_k, (long long)(E + N) * HEADS, alpha, d2, mbuf, (long long)(E + N));
  fill(sbuf, (long long)N * HEADS, 0.f);
  EW(expsum_k, (long long)(E + N) * HEADS, alpha, d2, mbuf, sbuf, (long long)(E + N));
  EW(segdiv_k, (long long)(E + N) * HEADS, alpha, d2, sbuf, (long long)(E + N));
  fill(hagg, (long long)N * 384, 0.f);
  EW(scatter3_k, (long long)(E + N) * HEADS * 64, xs, alpha, s2, d2, hagg, (long long)(E + N), 64);
  EW(addrow_k, (long long)N * 384, hagg, PF(70), (long long)N, 384);
  ln(hagg, PF(71), PF(72), hln, N, 384, 1);            // n3 + relu
  gemm(hln, 384, PF(73), PF(74), h3f, N, 64, 0);       // p3

  // ---------- global gate + classifier ----------
  pool_k<<<8, 256, 0, stream>>>(h3f, N, 64, hg);
  gemm(hg, 192, PF(75), PF(76), gate64, 1, 64, 2);     // gg sigmoid
  EW(hf_k, (long long)N * 65, h3f, gate64, x, xm, N);
  gemm(xm, 65, PF(77), PF(78), hagg, N, 64, 0);        // cls.l1
  ln(hagg, PF(79), PF(80), hagg, N, 64, 1);            // cls.n1 + relu
  gemm(hagg, 64, PF(81), PF(82), trb, N, 32, 0);       // cls.l2
  ln(trb, PF(83), PF(84), trb, N, 32, 1);              // cls.n2 + relu
  gemm(trb, 32, PF(85), PF(86), (float*)d_out, N, 3, 0); // cls.l3
  EW(logsoftmax3_k, N, (float*)d_out, N);
  (void)out_size;
#undef EW
#undef PF
}
